// SwiGLU_MLP_15058155339839
// MI455X (gfx1250) — 8-bit matrix operand order repaired and checked by reading, compiled, not run on hardware
//
#include <hip/hip_runtime.h>
#include <hip/hip_bf16.h>
#include <cstdint>


typedef int      v8i __attribute__((ext_vector_type(8)));
typedef _Float16 v8h __attribute__((ext_vector_type(8)));

#define D_MODEL 2048
#define D_FF    5504
#define TOKENS  8192
#define EPSF    1e-5f

__global__ void __launch_bounds__(256)
weight_quant_ternary(const float* __restrict__ w, int O, int I,
                     int8_t* __restrict__ wq, float* __restrict__ sc) {
    const int G = I >> 7;
    const int gid = blockIdx.x * (blockDim.x >> 5) + (threadIdx.x >> 5);
    if (gid >= O * G) return;
    const int lane = threadIdx.x & 31;
    const int o = gid / G, g = gid - o * G;

    const float* p = w + (size_t)o * I + g * 128 + lane * 4;
    float4 v = *(const float4*)p;
    float s = fabsf(v.x) + fabsf(v.y) + fabsf(v.z) + fabsf(v.w);
    #pragma unroll
    for (int off = 16; off > 0; off >>= 1) s += __shfl_xor(s, off, 32);
    const float scale = s * (1.0f / 128.0f) + EPSF;
    const float inv = 1.0f / scale;

    int b0 = (int)rintf(fminf(fmaxf(v.x * inv, -1.f), 1.f));
    int b1 = (int)rintf(fminf(fmaxf(v.y * inv, -1.f), 1.f));
    int b2 = (int)rintf(fminf(fmaxf(v.z * inv, -1.f), 1.f));
    int b3 = (int)rintf(fminf(fmaxf(v.w * inv, -1.f), 1.f));
    int packed = (b0 & 255) | ((b1 & 255) << 8) | ((b2 & 255) << 16) | ((b3 & 255) << 24);
    *(int*)(wq + (size_t)o * I + g * 128 + lane * 4) = packed;
    if (lane == 0) sc[(size_t)o * G + g] = scale;
}

__global__ void __launch_bounds__(256)
row_quant_f32(const float* __restrict__ src, int K,
              int8_t* __restrict__ q, float* __restrict__ dq) {
    const int row = blockIdx.x;
    const float4* s4 = (const float4*)(src + (size_t)row * K);
    const int n4 = K >> 2;
    float amax = 0.f;
    for (int i = threadIdx.x; i < n4; i += 256) {
        float4 v = s4[i];
        amax = fmaxf(amax, fmaxf(fmaxf(fabsf(v.x), fabsf(v.y)),
                                 fmaxf(fabsf(v.z), fabsf(v.w))));
    }
    __shared__ float red[256];
    red[threadIdx.x] = amax;
    __syncthreads();
    for (int off = 128; off > 0; off >>= 1) {
        if (threadIdx.x < off)
            red[threadIdx.x] = fmaxf(red[threadIdx.x], red[threadIdx.x + off]);
        __syncthreads();
    }
    amax = fmaxf(red[0], EPSF);
    const float s = 127.0f / amax;
    int* q4 = (int*)(q + (size_t)row * K);
    for (int i = threadIdx.x; i < n4; i += 256) {
        float4 v = s4[i];
        int b0 = max(-128, min(127, (int)rintf(v.x * s)));
        int b1 = max(-128, min(127, (int)rintf(v.y * s)));
        int b2 = max(-128, min(127, (int)rintf(v.z * s)));
        int b3 = max(-128, min(127, (int)rintf(v.w * s)));
        q4[i] = (b0 & 255) | ((b1 & 255) << 8) | ((b2 & 255) << 16) | ((b3 & 255) << 24);
    }
    if (threadIdx.x == 0) dq[row] = amax * (1.0f / 127.0f);
}

__global__ void __launch_bounds__(256)
row_quant_h16(const _Float16* __restrict__ src, int K,
              int8_t* __restrict__ q, float* __restrict__ dq) {
    const int row = blockIdx.x;
    const v8h* s8 = (const v8h*)(src + (size_t)row * K);
    const int n8 = K >> 3;
    float amax = 0.f;
    for (int i = threadIdx.x; i < n8; i += 256) {
        v8h v = s8[i];
        #pragma unroll
        for (int j = 0; j < 8; ++j) amax = fmaxf(amax, fabsf((float)v[j]));
    }
    __shared__ float red[256];
    red[threadIdx.x] = amax;
    __syncthreads();
    for (int off = 128; off > 0; off >>= 1) {
        if (threadIdx.x < off)
            red[threadIdx.x] = fmaxf(red[threadIdx.x], red[threadIdx.x + off]);
        __syncthreads();
    }
    amax = fmaxf(red[0], EPSF);
    const float s = 127.0f / amax;
    for (int i = threadIdx.x; i < n8; i += 256) {
        v8h v = s8[i];
        int lo = 0, hi = 0;
        #pragma unroll
        for (int j = 0; j < 4; ++j) {
            int b = max(-128, min(127, (int)rintf((float)v[j] * s)));
            lo |= (b & 255) << (8 * j);
        }
        #pragma unroll
        for (int j = 0; j < 4; ++j) {
            int b = max(-128, min(127, (int)rintf((float)v[4 + j] * s)));
            hi |= (b & 255) << (8 * j);
        }
        *(int2*)(q + (size_t)row * K + (size_t)i * 8) = make_int2(lo, hi);
    }
    if (threadIdx.x == 0) dq[row] = amax * (1.0f / 127.0f);
}

__device__ __forceinline__ v8i load_a_frag(const int8_t* arow, int k0, int h) {
    const int8_t* p = arow + k0 + h * 16;
    int2 t0 = *(const int2*)(p);
    int2 t1 = *(const int2*)(p + 8);
    int2 t2 = *(const int2*)(p + 32);
    int2 t3 = *(const int2*)(p + 40);
    v8i a;
    a[0] = t0.x; a[1] = t0.y; a[2] = t1.x; a[3] = t1.y;
    a[4] = t2.x; a[5] = t2.y; a[6] = t3.x; a[7] = t3.y;
    return a;
}
__device__ __forceinline__ v8i load_b_frag(const int8_t* brow, int k0, int h) {
    const int8_t* p = brow + k0 + h * 16;
    int4 t0 = *(const int4*)(p);
    int4 t1 = *(const int4*)(p + 32);
    v8i b;
    b[0] = t0.x; b[1] = t0.y; b[2] = t0.z; b[3] = t0.w;
    b[4] = t1.x; b[5] = t1.y; b[6] = t1.z; b[7] = t1.w;
    return b;
}

__device__ __forceinline__ v8i v8i_zero() {
    v8i z = {0, 0, 0, 0, 0, 0, 0, 0};
    return z;
}

__global__ void __launch_bounds__(256)
gemm1_gate_up_silu(const int8_t* __restrict__ xq, const float* __restrict__ xdq,
                   const int8_t* __restrict__ wg, const float* __restrict__ sg,
                   const int8_t* __restrict__ wu, const float* __restrict__ su,
                   _Float16* __restrict__ hidden) {
    const int K = D_MODEL, NG = K >> 7;
    const int lane = threadIdx.x & 31;
    const int wave = threadIdx.x >> 5;
    const int h = lane >> 4, nlo = lane & 15;
    const int n0 = blockIdx.x * 128 + wave * 16;
    const int m0 = blockIdx.y * 64;

    const int8_t* arow[4];
    #pragma unroll
    for (int t = 0; t < 4; ++t)
        arow[t] = xq + (size_t)(m0 + t * 16 + nlo) * K;
    const int8_t* bgr = wg + (size_t)(n0 + nlo) * K;
    const int8_t* bur = wu + (size_t)(n0 + nlo) * K;

    float fg[4][8], fu[4][8];
    #pragma unroll
    for (int t = 0; t < 4; ++t)
        #pragma unroll
        for (int j = 0; j < 8; ++j) { fg[t][j] = 0.f; fu[t][j] = 0.f; }

    for (int g = 0; g < NG; ++g) {
        if (g + 1 < NG) {
            __builtin_prefetch(bgr + ((g + 1) << 7) + h * 64, 0, 1);
            __builtin_prefetch(bur + ((g + 1) << 7) + h * 64, 0, 1);
        }
        v8i cg[4], cu[4];
        #pragma unroll
        for (int t = 0; t < 4; ++t) { cg[t] = v8i_zero(); cu[t] = v8i_zero(); }
        #pragma unroll
        for (int kk = 0; kk < 2; ++kk) {
            const int k0 = (g << 7) + (kk << 6);
            v8i bgf = load_b_frag(bgr, k0, h);
            v8i buf = load_b_frag(bur, k0, h);
            #pragma unroll
            for (int t = 0; t < 4; ++t) {
                v8i a = load_a_frag(arow[t], k0, h);
                cg[t] = __builtin_amdgcn_wmma_i32_16x16x64_iu8(true, a, true, bgf, cg[t], false, false);
                cu[t] = __builtin_amdgcn_wmma_i32_16x16x64_iu8(true, a, true, buf, cu[t], false, false);
            }
        }
        const float sgv = sg[(size_t)(n0 + nlo) * NG + g];
        const float suv = su[(size_t)(n0 + nlo) * NG + g];
        #pragma unroll
        for (int t = 0; t < 4; ++t)
            #pragma unroll
            for (int j = 0; j < 8; ++j) {
                fg[t][j] += (float)cg[t][j] * sgv;
                fu[t][j] += (float)cu[t][j] * suv;
            }
    }
    #pragma unroll
    for (int t = 0; t < 4; ++t)
        #pragma unroll
        for (int j = 0; j < 8; ++j) {
            const int m = m0 + t * 16 + j + 8 * h;
            const float dq = xdq[m];
            const float gv = fg[t][j] * dq;
            const float uv = fu[t][j] * dq;
            const float hv = (gv / (1.0f + __expf(-gv))) * uv;
            hidden[(size_t)m * D_FF + (n0 + nlo)] = (_Float16)hv;
        }
}

__global__ void __launch_bounds__(256)
gemm2_down(const int8_t* __restrict__ hq, const float* __restrict__ hdq,
           const int8_t* __restrict__ wd, const float* __restrict__ sd,
           float* __restrict__ out) {
    const int K = D_FF, NG = K >> 7;
    const int lane = threadIdx.x & 31;
    const int wave = threadIdx.x >> 5;
    const int h = lane >> 4, nlo = lane & 15;
    const int n0 = blockIdx.x * 128 + wave * 16;
    const int m0 = blockIdx.y * 64;

    const int8_t* arow[4];
    #pragma unroll
    for (int t = 0; t < 4; ++t)
        arow[t] = hq + (size_t)(m0 + t * 16 + nlo) * K;
    const int8_t* brow = wd + (size_t)(n0 + nlo) * K;

    float fa[4][8];
    #pragma unroll
    for (int t = 0; t < 4; ++t)
        #pragma unroll
        for (int j = 0; j < 8; ++j) fa[t][j] = 0.f;

    for (int g = 0; g < NG; ++g) {
        if (g + 1 < NG)
            __builtin_prefetch(brow + ((g + 1) << 7) + h * 64, 0, 1);
        v8i c[4];
        #pragma unroll
        for (int t = 0; t < 4; ++t) c[t] = v8i_zero();
        #pragma unroll
        for (int kk = 0; kk < 2; ++kk) {
            const int k0 = (g << 7) + (kk << 6);
            v8i b = load_b_frag(brow, k0, h);
            #pragma unroll
            for (int t = 0; t < 4; ++t) {
                v8i a = load_a_frag(arow[t], k0, h);
                c[t] = __builtin_amdgcn_wmma_i32_16x16x64_iu8(true, a, true, b, c[t], false, false);
            }
        }
        const float sv = sd[(size_t)(n0 + nlo) * NG + g];
        #pragma unroll
        for (int t = 0; t < 4; ++t)
            #pragma unroll
            for (int j = 0; j < 8; ++j) fa[t][j] += (float)c[t][j] * sv;
    }
    #pragma unroll
    for (int t = 0; t < 4; ++t)
        #pragma unroll
        for (int j = 0; j < 8; ++j) {
            const int m = m0 + t * 16 + j + 8 * h;
            out[(size_t)m * D_MODEL + (n0 + nlo)] = fa[t][j] * hdq[m];
        }
}

extern "C" void kernel_launch(void* const* d_in, const int* in_sizes, int n_in,
                              void* d_out, int out_size, void* d_ws, size_t ws_size,
                              hipStream_t stream) {
    const float* x      = (const float*)d_in[0];
    const float* w_gate = (const float*)d_in[1];
    const float* w_up   = (const float*)d_in[2];
    const float* w_down = (const float*)d_in[3];
    float* out = (float*)d_out;

    uint8_t* ws = (uint8_t*)d_ws;
    size_t off = 0;
    int8_t* wq_g = (int8_t*)(ws + off); off += (size_t)D_FF * D_MODEL;
    int8_t* wq_u = (int8_t*)(ws + off); off += (size_t)D_FF * D_MODEL;
    int8_t* wq_d = (int8_t*)(ws + off); off += (size_t)D_MODEL * D_FF;
    float* sc_g  = (float*)(ws + off);  off += (size_t)D_FF * 16 * 4;
    float* sc_u  = (float*)(ws + off);  off += (size_t)D_FF * 16 * 4;
    float* sc_d  = (float*)(ws + off);  off += (size_t)D_MODEL * 43 * 4;
    int8_t* xq   = (int8_t*)(ws + off); off += (size_t)TOKENS * D_MODEL;
    float* xdq   = (float*)(ws + off);  off += (size_t)TOKENS * 4;
    _Float16* hidden = (_Float16*)(ws + off); off += (size_t)TOKENS * D_FF * 2;
    int8_t* hq   = (int8_t*)(ws + off); off += (size_t)TOKENS * D_FF;
    float* hdq   = (float*)(ws + off);  off += (size_t)TOKENS * 4;
    (void)ws_size; (void)in_sizes; (void)n_in; (void)out_size;

    {
        int groups = D_FF * (D_MODEL / 128);
        int blocks = (groups + 7) / 8;
        weight_quant_ternary<<<blocks, 256, 0, stream>>>(w_gate, D_FF, D_MODEL, wq_g, sc_g);
        weight_quant_ternary<<<blocks, 256, 0, stream>>>(w_up,   D_FF, D_MODEL, wq_u, sc_u);
        weight_quant_ternary<<<blocks, 256, 0, stream>>>(w_down, D_MODEL, D_FF, wq_d, sc_d);
    }
    row_quant_f32<<<TOKENS, 256, 0, stream>>>(x, D_MODEL, xq, xdq);
    gemm1_gate_up_silu<<<dim3(D_FF / 128, TOKENS / 64), 256, 0, stream>>>(
        xq, xdq, wq_g, sc_g, wq_u, sc_u, hidden);
    row_quant_h16<<<TOKENS, 256, 0, stream>>>(hidden, D_FF, hq, hdq);
    gemm2_down<<<dim3(D_MODEL / 128, TOKENS / 64), 256, 0, stream>>>(
        hq, hdq, wq_d, sc_d, out);
}
